// StructuralSimilarityIndexMeasureMO_29755533426844
// MI455X (gfx1250) — compile-verified, matrix operand order checked by reading, not run on hardware
//
#include <hip/hip_runtime.h>

// SSIM via WMMA-accelerated separable 11x11 Gaussian convolution (gfx1250).
// 5 conv quantities (x, y, x^2, y^2, xy) -> fused SSIM map -> global mean.
//
// Per 128-thread block (4 waves): 32x32 output tile. Each wave owns a 16x16
// quadrant. Horizontal pass: D[16x16] = In[16x32] x Band[32x16] (two M-tiles
// to cover the 26-row vertical halo). Vertical pass: D = Band[16x32] x T[32x16]
// where T is repacked lane-locally from the horizontal D tiles (the f32 C/D
// layout and the f16 B layout assign the same (row set, column) to each lane).

typedef __attribute__((ext_vector_type(16))) _Float16 v16h;
typedef __attribute__((ext_vector_type(8)))  float    v8f;
typedef __attribute__((ext_vector_type(8)))  unsigned v8u;

#define IMG     512
#define TILE    32
#define HALO    5
#define REG     42            // TILE + 2*HALO (valid staged region)
#define QROWS   48            // padded rows (A-row index reaches 47)
#define QPITCH  25            // u32 pitch = 50 f16 per row (col index reaches 47)
#define QSZ     (QROWS * QPITCH)
#define WIN     11

__device__ __constant__ float GWC[WIN] = {
    0.0010284f, 0.0075988f, 0.0360008f, 0.1093607f, 0.2130055f,
    0.2660117f, 0.2130055f, 0.1093607f, 0.0360008f, 0.0075988f, 0.0010284f};

// K position of the low f16 half of VGPR v for 16-bit A/B fragments (wave32)
__device__ __forceinline__ int kmap(int v, int grp) {
    return ((v < 4) ? (2 * v) : (16 + 2 * (v - 4))) + 8 * grp;
}

__device__ __forceinline__ unsigned pkh2(float a, float b) {
    union { _Float16 h[2]; unsigned u; } u;
    u.h[0] = (_Float16)a;
    u.h[1] = (_Float16)b;
    return u.u;
}

__global__ void __launch_bounds__(128)
ssim_wmma_kernel(const float* __restrict__ X, const float* __restrict__ Y,
                 float* __restrict__ out, float invN)
{
    __shared__ unsigned sQ[5 * QSZ];   // five quantity images, f16 pairs
    __shared__ float    sWT[48];       // zero-padded Gaussian taps
    __shared__ float    sRed;

    const int tid  = threadIdx.x;
    const int lane = tid & 31;
    const int wave = tid >> 5;         // quadrant id 0..3
    const int qr   = wave >> 1;
    const int qc   = wave & 1;
    const int mn   = lane & 15;        // M (A rows) / N (B,C,D cols)
    const int grp  = lane >> 4;

    const int gr0 = blockIdx.y * TILE;
    const int gc0 = blockIdx.x * TILE;
    const float* Xs = X + (size_t)blockIdx.z * IMG * IMG;
    const float* Ys = Y + (size_t)blockIdx.z * IMG * IMG;

    if (tid == 0) sRed = 0.f;
    if (tid < 48) sWT[tid] = (tid >= 16 && tid <= 26) ? GWC[tid - 16] : 0.f;

    // ---- Stage: load x,y (clamped addr + select-zero -> no divergent loads),
    //      compute the 5 quantities, store packed f16 pairs, zero padding. ----
    for (int i = tid; i < QSZ; i += 128) {
        int r  = i / QPITCH;
        int c0 = 2 * (i % QPITCH);
        int R  = gr0 - HALO + r;
        int C  = gc0 - HALO + c0;
        int Rc  = min(max(R, 0), IMG - 1);
        int C0c = min(max(C, 0), IMG - 1);
        int C1c = min(max(C + 1, 0), IMG - 1);
        bool okr = ((unsigned)R < IMG) && (r < REG);
        bool ok0 = okr && ((unsigned)C < IMG) && (c0 < REG);
        bool ok1 = okr && ((unsigned)(C + 1) < IMG) && (c0 + 1 < REG);
        const float* xr = Xs + (size_t)Rc * IMG;
        const float* yr = Ys + (size_t)Rc * IMG;
        float xa = xr[C0c], xb = xr[C1c];
        float ya = yr[C0c], yb = yr[C1c];
        float x0 = ok0 ? xa : 0.f, x1 = ok1 ? xb : 0.f;
        float y0 = ok0 ? ya : 0.f, y1 = ok1 ? yb : 0.f;
        sQ[0 * QSZ + i] = pkh2(x0, x1);
        sQ[1 * QSZ + i] = pkh2(y0, y1);
        sQ[2 * QSZ + i] = pkh2(x0 * x0, x1 * x1);
        sQ[3 * QSZ + i] = pkh2(y0 * y0, y1 * y1);
        sQ[4 * QSZ + i] = pkh2(x0 * y0, x1 * y1);
    }
    __syncthreads();

    // ---- Banded Gaussian fragment: band[k][r] = g[k-r] (branch-free) ----
    // Serves as B in the horizontal pass and A in the vertical pass.
    v16h bandF;
#pragma unroll
    for (int v = 0; v < 8; ++v) {
        int k0 = kmap(v, grp);
        bandF[2 * v]     = (_Float16)sWT[k0 - mn + 16];
        bandF[2 * v + 1] = (_Float16)sWT[k0 + 1 - mn + 16];
    }

    const v8f zero = {};
    v8f acc[5];

#pragma unroll
    for (int q = 0; q < 5; ++q) {
        const unsigned* Q = &sQ[q * QSZ];

        // A fragments: two 16-row M tiles covering the 26 halo rows.
        // Each (k0,k0+1) f16 pair is one unconditional dword load.
        v16h a[2];
#pragma unroll
        for (int t = 0; t < 2; ++t) {
            int row = qr * 16 + t * 16 + mn;
            v8u au;
#pragma unroll
            for (int v = 0; v < 8; ++v) {
                int col = qc * 16 + kmap(v, grp);
                au[v] = Q[row * QPITCH + (col >> 1)];
            }
            a[t] = __builtin_bit_cast(v16h, au);
        }

        // Horizontal pass.
        v8f h0 = __builtin_amdgcn_wmma_f32_16x16x32_f16(
            false, a[0], false, bandF, (short)0, zero, false, false);
        v8f h1 = __builtin_amdgcn_wmma_f32_16x16x32_f16(
            false, a[1], false, bandF, (short)0, zero, false, false);

        // Lane-local repack: D rows {8g+v} / {16+8g+v} at col n are exactly
        // the B-fragment k positions for this lane. Zero T rows >= 26.
        v8u bu;
#pragma unroll
        for (int j = 0; j < 4; ++j) {
            bu[j] = pkh2(h0[2 * j], h0[2 * j + 1]);
            int r1 = 16 + 8 * grp + 2 * j;
            float e0 = (r1     <= 25) ? h1[2 * j]     : 0.f;
            float e1 = (r1 + 1 <= 25) ? h1[2 * j + 1] : 0.f;
            bu[4 + j] = pkh2(e0, e1);
        }
        v16h bT = __builtin_bit_cast(v16h, bu);

        // Vertical pass.
        acc[q] = __builtin_amdgcn_wmma_f32_16x16x32_f16(
            false, bandF, false, bT, (short)0, zero, false, false);
    }

    // ---- Fused SSIM map + mean reduction ----
    const float C1 = 1e-4f, C2 = 9e-4f;
    float partial = 0.f;
#pragma unroll
    for (int v = 0; v < 8; ++v) {
        float mx  = acc[0][v], my = acc[1][v];
        float mx2 = mx * mx, my2 = my * my, mxy = mx * my;
        float sxx = acc[2][v] - mx2;
        float syy = acc[3][v] - my2;
        float sxy = acc[4][v] - mxy;
        float num = (2.f * mxy + C1) * (2.f * sxy + C2);
        float den = (mx2 + my2 + C1) * (sxx + syy + C2);
        partial += num / den;
    }
    atomicAdd(&sRed, partial);
    __syncthreads();
    if (tid == 0) atomicAdd(out, sRed * invN);
}

__global__ void zero_out_kernel(float* o) { o[0] = 0.f; }

extern "C" void kernel_launch(void* const* d_in, const int* in_sizes, int n_in,
                              void* d_out, int out_size, void* d_ws, size_t ws_size,
                              hipStream_t stream) {
    const float* x = (const float*)d_in[0];
    const float* y = (const float*)d_in[1];
    float* out = (float*)d_out;

    long long n = in_sizes[0];                    // 32*3*512*512
    int slices = (int)(n / (long long)(IMG * IMG));
    float invN = 1.0f / (float)n;

    zero_out_kernel<<<1, 1, 0, stream>>>(out);
    dim3 grid(IMG / TILE, IMG / TILE, slices);
    ssim_wmma_kernel<<<grid, 128, 0, stream>>>(x, y, out, invN);
}
